// HeteroGNN_49074296324796
// MI455X (gfx1250) — matrix operand order repaired and checked by reading, not compiled, not run on hardware
//
#include <hip/hip_runtime.h>

// ---------------------------------------------------------------------------
// HeteroGNN (SAGEConv x6 edge types x3 layers + linear head) for gfx1250.
// fp32 scatter/normalize path + bf16 WMMA (v_wmma_f32_16x16x32_bf16) GEMMs.
// Weight (B) tiles are staged into LDS with the Tensor Data Mover
// (tensor_load_to_lds + s_wait_tensorcnt), A streams from global with
// global_prefetch_b8.
// ---------------------------------------------------------------------------

constexpr int FD   = 256;   // feature dim D
constexpr int FH   = 256;   // hidden H
constexpr int NL   = 3;     // layers
constexpr int NOUT = 1024;  // output dim

typedef __attribute__((ext_vector_type(16))) __bf16         v16bf;
typedef __attribute__((ext_vector_type(8)))  float          v8f;
typedef __attribute__((ext_vector_type(8)))  unsigned short v8us;
typedef __attribute__((ext_vector_type(4)))  unsigned int   v4ui;
typedef __attribute__((ext_vector_type(8)))  int            v8si;
typedef __attribute__((ext_vector_type(4)))  int            v4si;

union FragBF {
    struct { v8us lo, hi; } u;
    v16bf b;
};

// float -> bf16 (round to nearest even), stored as ushort
__device__ __forceinline__ unsigned short f2bf(float f) {
    unsigned u = __float_as_uint(f);
    unsigned r = u + 0x7FFFu + ((u >> 16) & 1u);
    return (unsigned short)(r >> 16);
}

// ------------------------------ element-wise -------------------------------

__global__ void k_zero(float* p, size_t n) {
    size_t i = (size_t)blockIdx.x * blockDim.x + threadIdx.x;
    if (i < n) p[i] = 0.0f;
}

__global__ void k_broadcast_row(float* dst, const float* __restrict__ row, size_t n) {
    size_t i = (size_t)blockIdx.x * blockDim.x + threadIdx.x;
    if (i < n) dst[i] = row[i & (FD - 1)];
}

__global__ void k_gather_rows(float* xp, const float* __restrict__ table,
                              const int* __restrict__ idx, size_t n) {
    size_t i = (size_t)blockIdx.x * blockDim.x + threadIdx.x;
    if (i < n) {
        size_t r = i >> 8;           // /FD
        int    f = (int)(i & (FD - 1));
        xp[i] = table[(size_t)idx[r] * FD + f];
    }
}

__global__ void k_relu(float* x, const float* __restrict__ acc, size_t n) {
    size_t i = (size_t)blockIdx.x * blockDim.x + threadIdx.x;
    if (i < n) x[i] = fmaxf(acc[i], 0.0f);
}

__global__ void k_relu_copy(float* x, float* embl, const float* __restrict__ acc, size_t n) {
    size_t i = (size_t)blockIdx.x * blockDim.x + threadIdx.x;
    if (i < n) {
        float v = fmaxf(acc[i], 0.0f);
        x[i]    = v;
        embl[i] = v;
    }
}

// ------------------------------ scatter-mean -------------------------------
// One wave32 per edge: 8 float atomics per lane covering 256 features.
__global__ void k_scatter_add(const float* __restrict__ xsrc,
                              const int* __restrict__ src, const int* __restrict__ dst,
                              float* s, float* cnt, int nedges) {
    int wave = blockIdx.x * (blockDim.x >> 5) + (threadIdx.x >> 5);
    int lane = threadIdx.x & 31;
    if (wave >= nedges) return;
    int si = src[wave], di = dst[wave];
    const float* xs = xsrc + (size_t)si * FD;
    float*       sd = s    + (size_t)di * FD;
#pragma unroll
    for (int i = 0; i < 8; ++i)
        atomicAdd(&sd[lane + i * 32], xs[lane + i * 32]);
    if (lane == 0) atomicAdd(&cnt[di], 1.0f);
}

// A = [ mean | x_dst ] in bf16, row-major [n, 512]
__global__ void k_build_A(unsigned short* A, const float* __restrict__ s,
                          const float* __restrict__ cnt, const float* __restrict__ xdst,
                          int n) {
    size_t tid = (size_t)blockIdx.x * blockDim.x + threadIdx.x;
    if (tid >= (size_t)n * FD) return;
    size_t row = tid >> 8;
    int    f   = (int)(tid & (FD - 1));
    float  c   = fmaxf(cnt[row], 1.0f);
    A[row * (2 * FD) + f]      = f2bf(s[tid] / c);
    A[row * (2 * FD) + FD + f] = f2bf(xdst[tid]);
}

// Wt[h][k] = k<256 ? Wl[k][h] : Wr[k-256][h]  (N-major bf16, [256][512])
__global__ void k_conv_W(unsigned short* Wt, const float* __restrict__ Wl,
                         const float* __restrict__ Wr) {
    int tid = blockIdx.x * blockDim.x + threadIdx.x;
    if (tid >= FH * 2 * FD) return;
    int h = tid >> 9;          // /512
    int k = tid & 511;
    float v = (k < FD) ? Wl[(size_t)k * FH + h] : Wr[(size_t)(k - FD) * FH + h];
    Wt[tid] = f2bf(v);
}

// Wout_t[o][k] = W_out[k][o]  ([1024][768] bf16)
__global__ void k_conv_Wout(unsigned short* Wt, const float* __restrict__ Wout) {
    int tid = blockIdx.x * blockDim.x + threadIdx.x;
    if (tid >= NOUT * (NL * FH)) return;
    int o = tid / (NL * FH);
    int k = tid - o * (NL * FH);
    Wt[tid] = f2bf(Wout[(size_t)k * NOUT + o]);
}

// A_final[i][l*256+f] = embL[l][i][f]   ([NR, 768] bf16)
__global__ void k_build_Afinal(unsigned short* A, const float* __restrict__ embL,
                               int nr) {
    size_t tid = (size_t)blockIdx.x * blockDim.x + threadIdx.x;
    if (tid >= (size_t)nr * NL * FH) return;
    size_t row = tid / (NL * FH);
    int    k   = (int)(tid - row * (NL * FH));
    int    l   = k >> 8;
    int    f   = k & (FH - 1);
    A[tid] = f2bf(embL[(size_t)l * nr * FH + row * FH + f]);
}

// --------------------------- WMMA bf16 GEMM --------------------------------
// C[n,N] = A[n,K](bf16) * Bt[N,K]^T(bf16) + bias[N]   (f32 accumulate)
// Block: 256 threads = 8 waves -> 128x64 C tile; wave -> 16x64 strip.
// B tile [64 x K] is DMA'd into LDS once per block by the Tensor Data Mover;
// A streams from global with prefetch.
__global__ __launch_bounds__(256) void k_gemm_bf16(
    const unsigned short* __restrict__ A,
    const unsigned short* __restrict__ Bt,
    const float* __restrict__ bias,
    float* __restrict__ C,
    int n, int N, int K) {
    extern __shared__ unsigned short Bs[];   // [64][K] bf16, dynamic LDS @ offset 0

    const int wave = threadIdx.x >> 5;
    const int lane = threadIdx.x & 31;
    const int hi   = lane >> 4;      // half-wave select
    const int lo   = lane & 15;      // M (A) / N (B,C) index within tile
    const int m0   = blockIdx.x * 128 + wave * 16;
    const int n0   = blockIdx.y * 64;

    // ---- TDM: load B tile Bt[n0 .. n0+63][0..K) -> LDS (issued by wave 0) ----
    if (threadIdx.x < 32) {
        unsigned long long ga = (unsigned long long)(Bt + (size_t)n0 * (size_t)K);
        v4ui g0;
        g0[0] = 1u;                                    // count=1, user descriptor
        g0[1] = 0u;                                    // lds_addr = 0 (dynamic LDS base)
        g0[2] = (unsigned)(ga & 0xFFFFFFFFu);          // global_addr[31:0]
        g0[3] = (unsigned)((ga >> 32) & 0x01FFFFFFu)   // global_addr[56:32]
              | (2u << 30);                            // type = 2 (image)
        v8si g1;
        g1[0] = (int)(1u << 16);                       // data_size=1 (2B), wg_mask=0
        g1[1] = (int)(((unsigned)K & 0xFFFFu) << 16);  // tensor_dim0[15:0]
        g1[2] = (int)(((unsigned)N & 0xFFFFu) << 16);  // tensor_dim1[15:0]
        g1[3] = (int)((((unsigned)K & 0xFFFFu) << 16)  // tile_dim0 = K
              | (((unsigned)N >> 16) & 0xFFFFu));      // tensor_dim1[31:16]
        g1[4] = 64;                                    // tile_dim1 = 64 rows
        g1[5] = K;                                     // tensor_dim0_stride[31:0]
        g1[6] = 0;
        g1[7] = 0;
        v4si gz = {0, 0, 0, 0};                        // groups 2/3 unused (2-D tile)
#if __clang_major__ >= 23
        v8si gz8 = {0, 0, 0, 0, 0, 0, 0, 0};
        __builtin_amdgcn_tensor_load_to_lds(g0, g1, gz, gz, gz8, 0);
#else
        __builtin_amdgcn_tensor_load_to_lds(g0, g1, gz, gz, 0);
#endif
        __builtin_amdgcn_s_wait_tensorcnt(0);
    }
    __syncthreads();

    int arow = m0 + lo;
    if (arow >= n) arow = n - 1;     // clamp (stores are guarded)
    const unsigned short* ap = A + (size_t)arow * K + hi * 8;

    v8f zero = {};
    v8f acc0 = zero, acc1 = zero, acc2 = zero, acc3 = zero;

    // B fragments now come from LDS (row-within-tile * K + K-chunk)
    const unsigned short* bs0 = Bs + (size_t)(0 * 16 + lo) * K + hi * 8;
    const unsigned short* bs1 = Bs + (size_t)(1 * 16 + lo) * K + hi * 8;
    const unsigned short* bs2 = Bs + (size_t)(2 * 16 + lo) * K + hi * 8;
    const unsigned short* bs3 = Bs + (size_t)(3 * 16 + lo) * K + hi * 8;

    for (int k = 0; k < K; k += 32) {
        // A fragment: 16x32 bf16, lane(hi,lo): K chunks [hi*8,+8) and [16+hi*8,+8)
        FragBF a;
        a.u.lo = *(const v8us*)(ap + k);
        a.u.hi = *(const v8us*)(ap + k + 16);
        __builtin_prefetch(ap + k + 64, 0, 1);   // global_prefetch_b8 for next A

        FragBF b0, b1, b2, b3;
        b0.u.lo = *(const v8us*)(bs0 + k);  b0.u.hi = *(const v8us*)(bs0 + k + 16);
        b1.u.lo = *(const v8us*)(bs1 + k);  b1.u.hi = *(const v8us*)(bs1 + k + 16);
        b2.u.lo = *(const v8us*)(bs2 + k);  b2.u.hi = *(const v8us*)(bs2 + k + 16);
        b3.u.lo = *(const v8us*)(bs3 + k);  b3.u.hi = *(const v8us*)(bs3 + k + 16);

        acc0 = __builtin_amdgcn_wmma_f32_16x16x32_bf16(false, a.b, false, b0.b,
                                                       (short)0, acc0, false, false);
        acc1 = __builtin_amdgcn_wmma_f32_16x16x32_bf16(false, a.b, false, b1.b,
                                                       (short)0, acc1, false, false);
        acc2 = __builtin_amdgcn_wmma_f32_16x16x32_bf16(false, a.b, false, b2.b,
                                                       (short)0, acc2, false, false);
        acc3 = __builtin_amdgcn_wmma_f32_16x16x32_bf16(false, a.b, false, b3.b,
                                                       (short)0, acc3, false, false);
    }

    v8f accs[4] = {acc0, acc1, acc2, acc3};
#pragma unroll
    for (int t = 0; t < 4; ++t) {
        int   col = n0 + t * 16 + lo;
        float bv  = bias ? bias[col] : 0.0f;
#pragma unroll
        for (int r = 0; r < 8; ++r) {
            int row = m0 + hi * 8 + r;     // C layout: c[r] = C[m0+hi*8+r][n0..+lo]
            if (row < n) C[(size_t)row * N + col] = accs[t][r] + bv;
        }
    }
}

// ----------------------- row L2-normalize + accumulate ---------------------
__global__ void k_norm_acc(const float* __restrict__ C, float* acc, int n) {
    int wave = blockIdx.x * (blockDim.x >> 5) + (threadIdx.x >> 5);
    int lane = threadIdx.x & 31;
    if (wave >= n) return;
    const float* c = C + (size_t)wave * FH;
    float v[8], ss = 0.0f;
#pragma unroll
    for (int i = 0; i < 8; ++i) { v[i] = c[lane + i * 32]; ss += v[i] * v[i]; }
#pragma unroll
    for (int m = 16; m > 0; m >>= 1) ss += __shfl_xor(ss, m, 32);
    float inv = 1.0f / fmaxf(sqrtf(ss), 1e-12f);
    float* a = acc + (size_t)wave * FH;
#pragma unroll
    for (int i = 0; i < 8; ++i) a[lane + i * 32] += v[i] * inv;
}

// ---------------------------------------------------------------------------

extern "C" void kernel_launch(void* const* d_in, const int* in_sizes, int n_in,
                              void* d_out, int out_size, void* d_ws, size_t ws_size,
                              hipStream_t stream) {
    const float* emb_r  = (const float*)d_in[0];
    const float* emb_c  = (const float*)d_in[1];
    const float* ptable = (const float*)d_in[2];
    const float* Wl     = (const float*)d_in[3];   // [L,6,D,H]
    const float* Wr     = (const float*)d_in[4];   // [L,6,D,H]
    const float* bconv  = (const float*)d_in[5];   // [L,6,H]
    const float* W_out  = (const float*)d_in[6];   // [L*H, OUT]
    const float* b_out  = (const float*)d_in[7];   // [OUT]
    const int*   idx_p  = (const int*)d_in[8];

    const int NP = in_sizes[8];          // #protein nodes
    const int NR = out_size / NOUT;      // #reaction nodes (output rows)
    const int NC = NR;                   // reference dispatch: n_complex == n_reaction

    // ---- workspace carve (256B aligned) ----
    char* w = (char*)d_ws;
    auto carve = [&](size_t bytes) -> void* {
        void* p = (void*)w;
        w += (bytes + 255) & ~(size_t)255;
        return p;
    };
    float* x_r   = (float*)carve((size_t)NR * FD * 4);
    float* x_c   = (float*)carve((size_t)NC * FD * 4);
    float* x_p   = (float*)carve((size_t)NP * FD * 4);
    float* acc_r = (float*)carve((size_t)NR * FH * 4);
    float* acc_c = (float*)carve((size_t)NC * FH * 4);
    float* acc_p = (float*)carve((size_t)NP * FH * 4);
    float* sbuf  = (float*)carve((size_t)NP * FD * 4);   // segment sums, reused as GEMM C
    float* cnt   = (float*)carve((size_t)NP * 4);
    float* embL  = (float*)carve((size_t)NL * NR * FH * 4);
    unsigned short* Abf    = (unsigned short*)carve((size_t)NP * 2 * FD * 2);
    unsigned short* Wbf    = (unsigned short*)carve((size_t)FH * 2 * FD * 2);
    unsigned short* Woutbf = (unsigned short*)carve((size_t)NOUT * NL * FH * 2);
    unsigned short* Afin   = (unsigned short*)carve((size_t)NR * NL * FH * 2);
    (void)ws_size; (void)n_in;

    const int T = 256;
    auto blocks = [](size_t n) { return (unsigned)((n + 255) / 256); };

    // ---- initial node features ----
    k_broadcast_row<<<blocks((size_t)NR * FD), T, 0, stream>>>(x_r, emb_r, (size_t)NR * FD);
    k_broadcast_row<<<blocks((size_t)NC * FD), T, 0, stream>>>(x_c, emb_c, (size_t)NC * FD);
    k_gather_rows<<<blocks((size_t)NP * FD), T, 0, stream>>>(x_p, ptable, idx_p, (size_t)NP * FD);

    struct ET { const float* xs; const float* xd; float* acc; int nd; int si; int di; };
    ET edges[6] = {
        { x_p, x_r, acc_r, NR,  9, 10 },   // p -> r
        { x_c, x_r, acc_r, NR, 11, 12 },   // c -> r
        { x_r, x_c, acc_c, NC, 13, 14 },   // r -> c
        { x_p, x_c, acc_c, NC, 15, 16 },   // p -> c
        { x_r, x_p, acc_p, NP, 17, 18 },   // r -> p
        { x_c, x_p, acc_p, NP, 19, 20 },   // c -> p
    };

    const size_t smemConv = (size_t)64 * (2 * FD) * 2;        // 64 x 512 bf16 = 64 KB
    const size_t smemHead = (size_t)64 * (NL * FH) * 2;       // 64 x 768 bf16 = 96 KB

    for (int l = 0; l < NL; ++l) {
        k_zero<<<blocks((size_t)NR * FH), T, 0, stream>>>(acc_r, (size_t)NR * FH);
        k_zero<<<blocks((size_t)NC * FH), T, 0, stream>>>(acc_c, (size_t)NC * FH);
        k_zero<<<blocks((size_t)NP * FH), T, 0, stream>>>(acc_p, (size_t)NP * FH);

        for (int j = 0; j < 6; ++j) {
            const ET& e  = edges[j];
            const int Ej = in_sizes[e.si];
            const int* src = (const int*)d_in[e.si];
            const int* dst = (const int*)d_in[e.di];

            k_zero<<<blocks((size_t)e.nd * FD), T, 0, stream>>>(sbuf, (size_t)e.nd * FD);
            k_zero<<<blocks((size_t)e.nd), T, 0, stream>>>(cnt, (size_t)e.nd);

            k_scatter_add<<<(Ej + 7) / 8, T, 0, stream>>>(e.xs, src, dst, sbuf, cnt, Ej);
            k_build_A<<<blocks((size_t)e.nd * FD), T, 0, stream>>>(Abf, sbuf, cnt, e.xd, e.nd);

            const size_t wofs = (size_t)(l * 6 + j) * FD * FH;
            k_conv_W<<<blocks((size_t)FH * 2 * FD), T, 0, stream>>>(Wbf, Wl + wofs, Wr + wofs);

            dim3 gg((e.nd + 127) / 128, FH / 64);
            k_gemm_bf16<<<gg, T, smemConv, stream>>>(Abf, Wbf,
                                                     bconv + (size_t)(l * 6 + j) * FH,
                                                     sbuf, e.nd, FH, 2 * FD);
            k_norm_acc<<<(e.nd + 7) / 8, T, 0, stream>>>(sbuf, e.acc, e.nd);
        }

        k_relu_copy<<<blocks((size_t)NR * FH), T, 0, stream>>>(
            x_r, embL + (size_t)l * NR * FH, acc_r, (size_t)NR * FH);
        k_relu<<<blocks((size_t)NC * FH), T, 0, stream>>>(x_c, acc_c, (size_t)NC * FH);
        k_relu<<<blocks((size_t)NP * FH), T, 0, stream>>>(x_p, acc_p, (size_t)NP * FH);
    }

    // ---- output head: [NR, 768] @ [768, 1024] + b_out ----
    k_build_Afinal<<<blocks((size_t)NR * NL * FH), T, 0, stream>>>(Afin, embL, NR);
    k_conv_Wout<<<blocks((size_t)NOUT * NL * FH), T, 0, stream>>>(Woutbf, W_out);
    dim3 go((NR + 127) / 128, NOUT / 64);
    k_gemm_bf16<<<go, T, smemHead, stream>>>(Afin, Woutbf, b_out, (float*)d_out,
                                             NR, NOUT, NL * FH);
}
